// NodeGTransformer_43181601194857
// MI455X (gfx1250) — matrix operand order repaired and checked by reading, not compiled, not run on hardware
//
#include <hip/hip_runtime.h>
#include <hip/hip_bf16.h>
#include <math.h>

// ---------- types ----------
typedef __bf16 bf;
typedef __attribute__((ext_vector_type(16))) __bf16 v16bf;
typedef __attribute__((ext_vector_type(8)))  __bf16 v8bf;
typedef __attribute__((ext_vector_type(4)))  __bf16 v4bf;
typedef __attribute__((ext_vector_type(8)))  float  v8f;

union FragBF { v16bf v; v8bf h[2]; };
union Bf16x8 { v8bf v; bf e[8]; };

static __device__ __forceinline__ bf f2bf(float f) {
    unsigned int u = __float_as_uint(f);
    unsigned int r = (u + 0x7FFFu + ((u >> 16) & 1u)) >> 16;
    unsigned short s = (unsigned short)r;
    return __builtin_bit_cast(bf, s);
}

#define LDS_STRIDE 40   // 32 + 8 pad (20 dwords/row -> conflict-free b128 LDS reads)
constexpr int NB = 256; // N dimension of every GEMM in this model

// ---------- generic WMMA bf16 GEMM: D = epilogue(A @ Bt^T) ----------
// Block tile 128(M) x 128(N), K-step 32. 8 wave32s: 4 (M) x 2 (N),
// each wave owns a 32x64 tile = 2x4 accumulators = 8 WMMA per K-step.
// A  : row-major M x K bf16 (per-batch stride sAb elements)
// Bt : row-major N x K bf16 (B transposed; per-batch stride sBb)
// flags: bit0 = SiLU, bit1 = mask before SiLU, bit2 = mask after SiLU
__global__ __launch_bounds__(256) void gemm_wmma_bf16(
    const bf* __restrict__ A,  unsigned long long sAb, int lda,
    const bf* __restrict__ Bt, unsigned long long sBb, int ldb,
    int M, int K,
    const float* __restrict__ bias,
    const float* __restrict__ addsrc, unsigned long long sAdd,
    const float* __restrict__ colScale, const float* __restrict__ colShift,
    const float* __restrict__ mask, unsigned long long sMk,
    int flags,
    float* __restrict__ outF, unsigned long long sOF,
    bf*    __restrict__ outB, unsigned long long sOB,
    bf*    __restrict__ outT, unsigned long long sOT, int ldT)
{
    __shared__ bf As[128 * LDS_STRIDE];
    __shared__ bf Bs[128 * LDS_STRIDE];

    const int b    = blockIdx.z;
    const int m0   = blockIdx.x * 128;
    const int n0   = blockIdx.y * 128;
    const int tid  = threadIdx.x;
    const int lane = tid & 31;
    const int wave = tid >> 5;
    const int wm   = (wave & 3) * 32;  // wave M offset within block tile
    const int wn   = (wave >> 2) * 64; // wave N offset within block tile

    A  += (unsigned long long)b * sAb;
    Bt += (unsigned long long)b * sBb;

    v8f acc[2][4] = {};

    // staging: both tiles are 128 rows x 32 cols bf16 -> 16 bf16 (32B) per thread
    const int srow = tid >> 1, sseg = (tid & 1) * 16;
    // fragment addressing
    const int fr  = lane & 15;
    const int fka = (lane >> 4) * 8;   // A: K halves interleaved (+0..+7, +16..+23)
    const int fkb = (lane >> 4) * 8;   // B: K halves interleaved (+0..+7, +16..+23)

    for (int k0 = 0; k0 < K; k0 += 32) {
        __syncthreads();
        {   // A tile -> LDS
            FragBF t;
            t.v = *(const v16bf*)(A + (unsigned long long)(m0 + srow) * lda + k0 + sseg);
            *(v8bf*)(&As[srow * LDS_STRIDE + sseg])     = t.h[0];
            *(v8bf*)(&As[srow * LDS_STRIDE + sseg + 8]) = t.h[1];
        }
        {   // B tile (transposed source) -> LDS
            FragBF t;
            t.v = *(const v16bf*)(Bt + (unsigned long long)(n0 + srow) * ldb + k0 + sseg);
            *(v8bf*)(&Bs[srow * LDS_STRIDE + sseg])     = t.h[0];
            *(v8bf*)(&Bs[srow * LDS_STRIDE + sseg + 8]) = t.h[1];
        }
        if (k0 + 32 < K) {  // global_prefetch_b8 for the next K tile
            __builtin_prefetch(A  + (unsigned long long)(m0 + srow) * lda + k0 + 32 + sseg, 0, 3);
            __builtin_prefetch(Bt + (unsigned long long)(n0 + srow) * ldb + k0 + 32 + sseg, 0, 3);
        }
        __syncthreads();

        FragBF af[2], bfg[4];
        #pragma unroll
        for (int i = 0; i < 2; ++i) {
            const bf* p = &As[(wm + i * 16 + fr) * LDS_STRIDE + fka];
            af[i].h[0] = *(const v8bf*)(p);
            af[i].h[1] = *(const v8bf*)(p + 16);
        }
        #pragma unroll
        for (int j = 0; j < 4; ++j) {
            const bf* p = &Bs[(wn + j * 16 + fr) * LDS_STRIDE + fkb];
            bfg[j].h[0] = *(const v8bf*)(p);
            bfg[j].h[1] = *(const v8bf*)(p + 16);
        }
        #pragma unroll
        for (int i = 0; i < 2; ++i)
            #pragma unroll
            for (int j = 0; j < 4; ++j)
                acc[i][j] = __builtin_amdgcn_wmma_f32_16x16x32_bf16(
                    false, af[i].v, false, bfg[j].v, (short)0, acc[i][j], false, false);
    }

    // ---------- epilogue ----------
    const bool doSilu   = (flags & 1) != 0;
    const bool maskPre  = (flags & 2) != 0;
    const bool maskPost = (flags & 4) != 0;
    const int col   = lane & 15;
    const int rbase = (lane >> 4) * 8;

    #pragma unroll
    for (int i = 0; i < 2; ++i) {
        #pragma unroll
        for (int j = 0; j < 4; ++j) {
            const int n = n0 + wn + j * 16 + col;
            const float bi  = bias     ? bias[n]     : 0.f;
            const float bns = colScale ? colScale[n] : 1.f;
            const float bnb = colShift ? colShift[n] : 0.f;
            float cv[8];
            #pragma unroll
            for (int e = 0; e < 8; ++e) {
                const int m = m0 + wm + i * 16 + rbase + e;
                float c = acc[i][j][e];
                c += bi;
                if (addsrc)   c += addsrc[b * sAdd + (unsigned long long)m * NB + n];
                if (colScale) c = c * bns + bnb;
                float mk = 1.f;
                if (maskPre || maskPost) mk = mask[b * sMk + m];
                if (maskPre)  c *= mk;
                if (doSilu)   c = c * (1.f / (1.f + __expf(-c)));
                if (maskPost) c *= mk;
                cv[e] = c;
                if (outF) outF[b * sOF + (unsigned long long)m * NB + n] = c;
                if (outB) outB[b * sOB + (unsigned long long)m * NB + n] = f2bf(c);
            }
            if (outT) {
                Bf16x8 p;
                #pragma unroll
                for (int e = 0; e < 8; ++e) p.e[e] = f2bf(cv[e]);
                const int m = m0 + wm + i * 16 + rbase;
                *(v8bf*)(outT + b * sOT + (unsigned long long)n * ldT + m) = p.v;
            }
        }
    }
}

// ---------- helpers ----------
__global__ __launch_bounds__(256) void cvt_f32_bf16(
    const float* __restrict__ s, bf* __restrict__ d, unsigned long long n)
{
    unsigned long long i = ((unsigned long long)blockIdx.x * 256 + threadIdx.x) * 4;
    if (i + 3 < n) {
        const float4 f = *(const float4*)(s + i);
        union { v4bf v; bf e[4]; } p;
        p.e[0] = f2bf(f.x); p.e[1] = f2bf(f.y); p.e[2] = f2bf(f.z); p.e[3] = f2bf(f.w);
        *(v4bf*)(d + i) = p.v;
    }
}

// W (256x256, k-major) -> Wt (n-major, bf16)
__global__ __launch_bounds__(256) void cvtT256(const float* __restrict__ W, bf* __restrict__ Wt)
{
    const int k = blockIdx.x, n = threadIdx.x;
    Wt[(unsigned long long)n * 256 + k] = f2bf(W[(unsigned long long)k * 256 + n]);
}

__global__ void bnfold(const float* __restrict__ g, const float* __restrict__ be,
                       const float* __restrict__ mu, const float* __restrict__ var,
                       float* __restrict__ g2, float* __restrict__ b2)
{
    const int i = threadIdx.x;
    const float s = g[i] * rsqrtf(var[i] + 1e-3f);
    g2[i] = s;
    b2[i] = be[i] - mu[i] * s;
}

// ---------- host ----------
extern "C" void kernel_launch(void* const* d_in, const int* in_sizes, int n_in,
                              void* d_out, int out_size, void* d_ws, size_t ws_size,
                              hipStream_t stream)
{
    const float* x      = (const float*)d_in[0];
    const float* adj    = (const float*)d_in[1];
    const float* mask   = (const float*)d_in[2];
    const float* W_lin  = (const float*)d_in[3];
    const float* b_lin  = (const float*)d_in[4];
    const float* Wq_mp  = (const float*)d_in[5];
    const float* bq_mp  = (const float*)d_in[6];
    const float* Wk_mp  = (const float*)d_in[7];
    const float* bk_mp  = (const float*)d_in[8];
    const float* Wv_mp  = (const float*)d_in[9];
    const float* bv_mp  = (const float*)d_in[10];
    const float* Wk_att = (const float*)d_in[11];
    const float* Wv_att = (const float*)d_in[12];
    const float* Wq_att = (const float*)d_in[13];
    const float* Wo_att = (const float*)d_in[14];
    const float* bng    = (const float*)d_in[15];
    const float* bnb_   = (const float*)d_in[16];
    const float* bnm    = (const float*)d_in[17];
    const float* bnv    = (const float*)d_in[18];
    const float* W_proj = (const float*)d_in[19];
    const float* b_proj = (const float*)d_in[20];
    const float* W_res  = (const float*)d_in[21];
    const float* b_res  = (const float*)d_in[22];

    char* ws = (char*)d_ws;
    size_t off = 0;
    auto alloc = [&](size_t bytes) -> char* {
        char* p = ws + off;
        off += (bytes + 255) & ~(size_t)255;
        return p;
    };
    bf* adjb   = (bf*)alloc(67108864ULL);       // adj bf16 (8x2048x2048)
    bf* wt     = (bf*)alloc(16ULL * 65536 * 2); // 16 transposed weights
    bf* xb     = (bf*)alloc(8388608ULL);        // x bf16 row-major (16384x256)
    bf* ht     = (bf*)alloc(8388608ULL);        // h transposed [256][16384]
    bf* A1     = (bf*)alloc(8388608ULL);        // xn / pq row-major
    bf* T1     = (bf*)alloc(8388608ULL);        // ping transposed / pkt
    bf* T2     = (bf*)alloc(8388608ULL);        // pong transposed / pvt
    bf* qb     = (bf*)alloc(8388608ULL);
    bf* kb     = (bf*)alloc(8388608ULL);
    bf* vb     = (bf*)alloc(8388608ULL);
    bf* kvt    = (bf*)alloc(1048576ULL);        // kv^T per batch [e][d]
    float* projf = (float*)alloc(16777216ULL);  // proj output f32
    float* g2  = (float*)alloc(1024);
    float* b2  = (float*)alloc(1024);
    (void)in_sizes; (void)n_in; (void)out_size; (void)ws_size;

    // conversions
    cvt_f32_bf16<<<dim3(33554432u / 1024u), 256, 0, stream>>>(adj, adjb, 33554432ULL);
    cvt_f32_bf16<<<dim3(4194304u  / 1024u), 256, 0, stream>>>(x,   xb,   4194304ULL);
    const float* wsrc[16] = {
        W_lin,
        Wq_mp, Wq_mp + 65536, Wq_mp + 131072,
        Wk_mp, Wk_mp + 65536, Wk_mp + 131072,
        Wv_mp, Wv_mp + 65536, Wv_mp + 131072,
        Wk_att, Wv_att, Wq_att, Wo_att, W_proj, W_res };
    for (int i = 0; i < 16; ++i)
        cvtT256<<<256, 256, 0, stream>>>(wsrc[i], wt + (size_t)i * 65536);
    bnfold<<<1, 256, 0, stream>>>(bng, bnb_, bnm, bnv, g2, b2);

    auto G = [&](dim3 grid,
                 const bf* Ap, unsigned long long sAb, int lda,
                 const bf* Btp, unsigned long long sBb, int ldb,
                 int M, int K,
                 const float* bias_,
                 const float* addsrc, unsigned long long sAdd,
                 const float* cS, const float* cSh,
                 const float* mk, unsigned long long sMk, int flags,
                 float* oF, unsigned long long sOF,
                 bf* oB, unsigned long long sOB,
                 bf* oT, unsigned long long sOT, int ldT) {
        gemm_wmma_bf16<<<grid, 256, 0, stream>>>(Ap, sAb, lda, Btp, sBb, ldb, M, K,
            bias_, addsrc, sAdd, cS, cSh, mk, sMk, flags,
            oF, sOF, oB, sOB, oT, sOT, ldT);
    };

    const dim3 flatG(128, 2, 1);   // M=16384, N=256
    const dim3 adjG (16,  2, 8);   // per-batch M=2048, K=2048
    const dim3 kvG  (2,   2, 8);   // per-batch M=256,  K=2048
    const dim3 attG (16,  2, 8);   // per-batch M=2048, K=256

    // h = x @ W_lin + b_lin   -> ht (transposed [256][16384])
    G(flatG, xb, 0, 256, wt + 0, 0, 256, 16384, 256,
      b_lin, nullptr, 0, nullptr, nullptr, nullptr, 0, 0,
      nullptr, 0, nullptr, 0, ht, 0, 16384);

    // dynamic message passing for q, k, v
    const float* mpB[3] = { bq_mp, bk_mp, bv_mp };
    bf* qkv[3] = { qb, kb, vb };
    for (int br = 0; br < 3; ++br) {
        const bf* cur = ht;
        for (int j = 0; j < 6; ++j) {
            // xn = adj @ cur  (cur consumed in transposed layout)
            G(adjG, adjb, 4194304ULL, 2048, cur, 2048ULL, 16384, 2048, 2048,
              nullptr, nullptr, 0, nullptr, nullptr, nullptr, 0, 0,
              nullptr, 0, A1, 524288ULL, nullptr, 0, 0);
            const int si = j >> 1;
            const bf* wslot = wt + (size_t)(1 + br * 3 + si) * 65536;
            const float* bias_ = mpB[br] + si * 256;
            if (j < 5) {
                bf* dst = (j & 1) ? T2 : T1;
                G(flatG, A1, 0, 256, wslot, 0, 256, 16384, 256,
                  bias_, nullptr, 0, nullptr, nullptr, nullptr, 0, /*silu*/1,
                  nullptr, 0, nullptr, 0, dst, 0, 16384);
                cur = dst;
            } else {
                // last hop: silu then * mask, row-major bf16 out
                G(flatG, A1, 0, 256, wslot, 0, 256, 16384, 256,
                  bias_, nullptr, 0, nullptr, nullptr, mask, 0, 1 | 4,
                  nullptr, 0, qkv[br], 0, nullptr, 0, 0);
            }
        }
    }

    // pk = silu((k @ Wk_att) * m)  -> T1 = pk^T
    G(flatG, kb, 0, 256, wt + 10 * 65536, 0, 256, 16384, 256,
      nullptr, nullptr, 0, nullptr, nullptr, mask, 0, 1 | 2,
      nullptr, 0, nullptr, 0, T1, 0, 16384);
    // pv = (v @ Wv_att) * m        -> T2 = pv^T
    G(flatG, vb, 0, 256, wt + 11 * 65536, 0, 256, 16384, 256,
      nullptr, nullptr, 0, nullptr, nullptr, mask, 0, 2,
      nullptr, 0, nullptr, 0, T2, 0, 16384);
    // pq = q @ Wq_att              -> A1 (row-major)
    G(flatG, qb, 0, 256, wt + 12 * 65536, 0, 256, 16384, 256,
      nullptr, nullptr, 0, nullptr, nullptr, nullptr, 0, 0,
      nullptr, 0, A1, 0, nullptr, 0, 0);
    // kv[b] = pk^T @ pv            -> kvt[b] = kv^T ([e][d])
    G(kvG, T1, 2048ULL, 16384, T2, 2048ULL, 16384, 256, 2048,
      nullptr, nullptr, 0, nullptr, nullptr, nullptr, 0, 0,
      nullptr, 0, nullptr, 0, kvt, 65536ULL, 256);
    // attn = pq @ kv               -> kb (reuse)
    G(attG, A1, 524288ULL, 256, kvt, 65536ULL, 256, 2048, 256,
      nullptr, nullptr, 0, nullptr, nullptr, nullptr, 0, 0,
      nullptr, 0, kb, 524288ULL, nullptr, 0, 0);
    // y = BN(attn @ Wo_att + x)    -> vb (reuse)
    G(flatG, kb, 0, 256, wt + 13 * 65536, 0, 256, 16384, 256,
      nullptr, x, 0, g2, b2, nullptr, 0, 0,
      nullptr, 0, vb, 0, nullptr, 0, 0);
    // proj = silu(y @ W_proj + b_proj) -> projf (f32)
    G(flatG, vb, 0, 256, wt + 14 * 65536, 0, 256, 16384, 256,
      b_proj, nullptr, 0, nullptr, nullptr, nullptr, 0, 1,
      projf, 0, nullptr, 0, nullptr, 0, 0);
    // out = (proj + x @ W_res + b_res) * mask -> d_out (f32)
    G(flatG, xb, 0, 256, wt + 15 * 65536, 0, 256, 16384, 256,
      b_res, projf, 0, nullptr, nullptr, mask, 0, 4,
      (float*)d_out, 0, nullptr, 0, nullptr, 0, 0);
}
